// ResidualMambaBlock_83958020702546
// MI455X (gfx1250) — matrix operand order repaired and checked by reading, not compiled, not run on hardware
//
#include <hip/hip_runtime.h>
#include <hip/hip_bf16.h>

// ---------------------------------------------------------------------------
// Mamba block for MI455X (gfx1250):
//  - GEMMs via v_wmma_f32_16x16x32_f16, 128x64 block tile, 8 waves,
//    double-buffered LDS fed by global_load_async_to_lds_b128 (ASYNCcnt).
//  - selective scan: 1 thread per (b,d), state in registers, B/C via LDS
//    with double buffering (1 barrier / timestep).
// ---------------------------------------------------------------------------

typedef __attribute__((ext_vector_type(16))) _Float16 v16h;
typedef __attribute__((ext_vector_type(8)))  float    v8f;

#define B_SZ     2
#define L_SZ     2048
#define D_MODEL  1024
#define D_INNER  2048
#define D_STATE  16
#define D_CONV   4
#define DT_RANK  64
#define XPROJ_N  (DT_RANK + 2 * D_STATE)   // 96
#define M_TOK    (B_SZ * L_SZ)             // 4096

__device__ __forceinline__ float sanitize_f(float v) {
    if (isnan(v)) return 0.0f;
    if (isinf(v)) return v > 0.0f ? 1e4f : -1e4f;
    return v;
}
__device__ __forceinline__ float sigmoid_f(float x) {
    return 1.0f / (1.0f + __expf(-x));
}
__device__ __forceinline__ float softplus_f(float x) {
    return x > 20.0f ? x : log1pf(__expf(x));
}

// low 32 bits of a generic shared-aperture address == LDS byte offset (ISA 10.2)
__device__ __forceinline__ unsigned lds_addr32(const void* p) {
    return (unsigned)(unsigned long long)(uintptr_t)p;
}
// CDNA5 async DMA: global -> LDS, 16B per lane, tracked by ASYNCcnt
__device__ __forceinline__ void async_ld_b128(unsigned lds_off, const void* gptr) {
    asm volatile("global_load_async_to_lds_b128 %0, %1, off"
                 :: "v"(lds_off), "v"((unsigned long long)(uintptr_t)gptr)
                 : "memory");
}
__device__ __forceinline__ void wait_asynccnt0() {
    asm volatile("s_wait_asynccnt 0x0" ::: "memory");
}

// ------------------------- f32 -> f16 convert ------------------------------
__global__ __launch_bounds__(256) void cvt_f32_f16(const float* __restrict__ src,
                                                   _Float16* __restrict__ dst, int n) {
    int i = blockIdx.x * 256 + threadIdx.x;
    if (i < n) dst[i] = (_Float16)src[i];
}

// ------------------------- LayerNorm (token per block) ---------------------
__global__ __launch_bounds__(256) void layernorm_k(const float* __restrict__ x,
                                                   const float* __restrict__ g,
                                                   const float* __restrict__ b,
                                                   _Float16* __restrict__ xn) {
    __shared__ float red[256];
    const int t = blockIdx.x;
    const float* xr = x + (size_t)t * D_MODEL;
    float v[4]; float s = 0.f;
#pragma unroll
    for (int i = 0; i < 4; ++i) { v[i] = sanitize_f(xr[threadIdx.x + 256 * i]); s += v[i]; }
    red[threadIdx.x] = s; __syncthreads();
    for (int o = 128; o > 0; o >>= 1) {
        if (threadIdx.x < o) red[threadIdx.x] += red[threadIdx.x + o];
        __syncthreads();
    }
    const float mu = red[0] * (1.0f / D_MODEL);
    __syncthreads();
    float s2 = 0.f;
#pragma unroll
    for (int i = 0; i < 4; ++i) { float d = v[i] - mu; s2 += d * d; }
    red[threadIdx.x] = s2; __syncthreads();
    for (int o = 128; o > 0; o >>= 1) {
        if (threadIdx.x < o) red[threadIdx.x] += red[threadIdx.x + o];
        __syncthreads();
    }
    const float r = rsqrtf(red[0] * (1.0f / D_MODEL) + 1e-5f);
#pragma unroll
    for (int i = 0; i < 4; ++i) {
        int c = threadIdx.x + 256 * i;
        xn[(size_t)t * D_MODEL + c] = (_Float16)((v[i] - mu) * r * g[c] + b[c]);
    }
}

// ------------------------- WMMA GEMM: C[M,ldc] = A[M,K] * W[N,K]^T ---------
// 256 threads = 8 waves. Block tile 128x64, wave tile 32x32 (4 accumulators).
// Double-buffered LDS, tiles DMA'd in with global_load_async_to_lds_b128.
#define TM 128
#define TN 64
#define TK 32
__global__ __launch_bounds__(256) void gemm_wmma_f16(const _Float16* __restrict__ A,
                                                     const _Float16* __restrict__ W,
                                                     float* __restrict__ C,
                                                     int N, int K, int ldc) {
    __shared__ __align__(16) _Float16 lA[2][TM * TK];   // 2 x 8 KB
    __shared__ __align__(16) _Float16 lB[2][TN * TK];   // 2 x 4 KB
    const int tid  = threadIdx.x;
    const int lane = tid & 31;
    const int wave = tid >> 5;
    const int wm   = wave & 3;     // 32-row group 0..3
    const int wn   = wave >> 2;    // 32-col group 0..1
    const int m0   = blockIdx.y * TM;
    const int n0   = blockIdx.x * TN;

    // per-thread DMA chunk coords: A tile = 512 b128 chunks (2/thread), B = 256 (1/thread)
    const int arow0 = tid >> 2;                 // chunks tid, tid+256 -> rows tid>>2, 64+(tid>>2)
    const int acol  = (tid & 3) * 8;            // halves
    const int brow  = tid >> 2;                 // 0..63
    const bool bvalid = (n0 + brow) < N;

    v8f acc[2][2] = {{{}, {}}, {{}, {}}};

    // ---- prologue: zero OOB weight rows (stay zero forever), DMA tile 0 ----
    if (!bvalid) {
        float4 z{};
        *(float4*)&lB[0][brow * TK + acol] = z;
        *(float4*)&lB[1][brow * TK + acol] = z;
    }
    async_ld_b128(lds_addr32(&lA[0][arow0 * TK + acol]),
                  A + (size_t)(m0 + arow0) * K + acol);
    async_ld_b128(lds_addr32(&lA[0][(64 + arow0) * TK + acol]),
                  A + (size_t)(m0 + 64 + arow0) * K + acol);
    if (bvalid)
        async_ld_b128(lds_addr32(&lB[0][brow * TK + acol]),
                      W + (size_t)(n0 + brow) * K + acol);
    wait_asynccnt0();
    __syncthreads();

    for (int k0 = 0; k0 < K; k0 += TK) {
        const int cur = (k0 / TK) & 1;
        const int nxt = cur ^ 1;
        const int k1  = k0 + TK;
        if (k1 < K) {   // DMA next tile while computing this one
            async_ld_b128(lds_addr32(&lA[nxt][arow0 * TK + acol]),
                          A + (size_t)(m0 + arow0) * K + k1 + acol);
            async_ld_b128(lds_addr32(&lA[nxt][(64 + arow0) * TK + acol]),
                          A + (size_t)(m0 + 64 + arow0) * K + k1 + acol);
            if (bvalid)
                async_ld_b128(lds_addr32(&lB[nxt][brow * TK + acol]),
                              W + (size_t)(n0 + brow) * K + k1 + acol);
        }

        // A fragments (16x32 f16, ISA 7.12.2): lanes 0-15 K{0-7,16-23}, 16-31 K{8-15,24-31}
        const int ksel = (lane >> 4) << 3;
        v16h af[2];
#pragma unroll
        for (int s = 0; s < 2; ++s) {
            const int am = (wm << 5) + (s << 4) + (lane & 15);
#pragma unroll
            for (int v = 0; v < 8; ++v) {
                int kk = ((v < 4) ? 0 : 16) + ksel + ((v & 3) << 1);
                af[s][2 * v]     = lA[cur][am * TK + kk];
                af[s][2 * v + 1] = lA[cur][am * TK + kk + 1];
            }
        }
        // B fragments (32x16 f16): lanes 0-15 K{0-7,16-23}, 16-31 K{8-15,24-31}
        const int bko = (lane >> 4) << 4;
        v16h bf[2];
#pragma unroll
        for (int t = 0; t < 2; ++t) {
            const int bn = (wn << 5) + (t << 4) + (lane & 15);
#pragma unroll
            for (int v = 0; v < 8; ++v) {
                int kk = ((v < 4) ? 0 : 16) + (bko >> 1) + ((v & 3) << 1);
                bf[t][2 * v]     = lB[cur][bn * TK + kk];
                bf[t][2 * v + 1] = lB[cur][bn * TK + kk + 1];
            }
        }
#pragma unroll
        for (int s = 0; s < 2; ++s)
#pragma unroll
            for (int t = 0; t < 2; ++t)
                acc[s][t] = __builtin_amdgcn_wmma_f32_16x16x32_f16(
                    false, af[s], false, bf[t], (short)0, acc[s][t], false, false);

        wait_asynccnt0();      // next tile landed in LDS
        __syncthreads();       // all waves done reading cur, next tile visible
    }

    // C layout: lane n = lane%16; VGPR r -> m = r + 8*(lane>=16)
    const int cn = lane & 15;
    const int mo = (lane >> 4) << 3;
#pragma unroll
    for (int s = 0; s < 2; ++s) {
#pragma unroll
        for (int t = 0; t < 2; ++t) {
            const int gn = n0 + (wn << 5) + (t << 4) + cn;
            if (gn >= N) continue;
#pragma unroll
            for (int r = 0; r < 8; ++r) {
                const int gm = m0 + (wm << 5) + (s << 4) + mo + r;
                C[(size_t)gm * ldc + gn] = acc[s][t][r];
            }
        }
    }
}

// ------------------------- causal depthwise conv + SiLU --------------------
__global__ __launch_bounds__(256) void conv_silu_k(const float* __restrict__ xz,
                                                   const float* __restrict__ cw,
                                                   const float* __restrict__ cb,
                                                   float* __restrict__ uconv,
                                                   _Float16* __restrict__ u16) {
    const int idx = blockIdx.x * 256 + threadIdx.x;   // over M*2048
    const int d  = idx & (D_INNER - 1);
    const int bl = idx >> 11;
    const int l  = bl & (L_SZ - 1);
    const int b  = bl >> 11;
    float acc = cb[d];
#pragma unroll
    for (int j = 0; j < D_CONV; ++j) {
        const int lj = l - (D_CONV - 1) + j;
        if (lj >= 0)
            acc += cw[d * D_CONV + j] * xz[((size_t)(b * L_SZ + lj)) * (2 * D_INNER) + d];
    }
    const float s = acc * sigmoid_f(acc);
    uconv[idx] = s;
    u16[idx]   = (_Float16)s;
}

// ------------------------- extract dt cols (f32[.,96] -> f16[.,64]) --------
__global__ __launch_bounds__(256) void extract_dt_k(const float* __restrict__ xdbl,
                                                    _Float16* __restrict__ dt16) {
    const int idx = blockIdx.x * 256 + threadIdx.x;   // over M*64
    const int r = idx >> 6, c = idx & 63;
    dt16[idx] = (_Float16)xdbl[(size_t)r * XPROJ_N + c];
}

// ------------------------- selective scan (softplus fused) -----------------
__global__ __launch_bounds__(256) void scan_k(const float* __restrict__ uconv,
                                              const float* __restrict__ draw,   // pre-softplus dt
                                              const float* __restrict__ dt_b,
                                              const float* __restrict__ xdbl,   // B/C at cols 64..95
                                              const float* __restrict__ A_log,
                                              const float* __restrict__ Dskip,
                                              float* __restrict__ y) {
    const int b = blockIdx.y;
    const int d = blockIdx.x * 256 + threadIdx.x;
    float Ad[D_STATE], h[D_STATE];
#pragma unroll
    for (int n = 0; n < D_STATE; ++n) { Ad[n] = -__expf(A_log[d * D_STATE + n]); h[n] = 0.f; }
    const float Dd = Dskip[d];
    const float db = dt_b[d];
    __shared__ float sBC[2][2 * D_STATE];
    if (threadIdx.x < 2 * D_STATE)
        sBC[0][threadIdx.x] = xdbl[((size_t)b * L_SZ) * XPROJ_N + DT_RANK + threadIdx.x];
    __syncthreads();
    for (int t = 0; t < L_SZ; ++t) {
        const int cur = t & 1;
        const size_t row = (size_t)b * L_SZ + t;
        if (threadIdx.x < 2 * D_STATE && t + 1 < L_SZ)
            sBC[cur ^ 1][threadIdx.x] = xdbl[(row + 1) * XPROJ_N + DT_RANK + threadIdx.x];
        const float ut = uconv[row * D_INNER + d];
        const float dt = softplus_f(draw[row * D_INNER + d] + db);
        const float du = dt * ut;
        float acc = 0.f;
#pragma unroll
        for (int n = 0; n < D_STATE; ++n) {
            h[n] = __expf(dt * Ad[n]) * h[n] + du * sBC[cur][n];
            acc += h[n] * sBC[cur][D_STATE + n];
        }
        y[row * D_INNER + d] = acc + ut * Dd;
        __syncthreads();
    }
}

// ------------------------- gate: yg = y * silu(z) --------------------------
__global__ __launch_bounds__(256) void gate_k(const float* __restrict__ y,
                                              const float* __restrict__ xz,
                                              _Float16* __restrict__ yg16) {
    const int idx = blockIdx.x * 256 + threadIdx.x;   // over M*2048
    const int d = idx & (D_INNER - 1);
    const int bl = idx >> 11;
    const float z = xz[(size_t)bl * (2 * D_INNER) + D_INNER + d];
    yg16[idx] = (_Float16)(y[idx] * (z * sigmoid_f(z)));
}

// ------------------------- residual + sanitize -----------------------------
__global__ __launch_bounds__(256) void resid_k(const float* __restrict__ x,
                                               const float* __restrict__ go,
                                               float* __restrict__ out, int n) {
    const int idx = blockIdx.x * 256 + threadIdx.x;
    if (idx < n) out[idx] = sanitize_f(sanitize_f(x[idx]) + sanitize_f(go[idx]));
}

// ---------------------------------------------------------------------------
extern "C" void kernel_launch(void* const* d_in, const int* in_sizes, int n_in,
                              void* d_out, int out_size, void* d_ws, size_t ws_size,
                              hipStream_t stream) {
    const float* x      = (const float*)d_in[0];
    const float* ln_g   = (const float*)d_in[1];
    const float* ln_b   = (const float*)d_in[2];
    const float* w_in   = (const float*)d_in[3];   // [4096,1024]
    const float* conv_w = (const float*)d_in[4];   // [2048,4]
    const float* conv_b = (const float*)d_in[5];
    const float* w_xp   = (const float*)d_in[6];   // [96,2048]
    const float* w_dt   = (const float*)d_in[7];   // [2048,64]
    const float* dt_b   = (const float*)d_in[8];
    const float* A_log  = (const float*)d_in[9];   // [2048,16]
    const float* Dskip  = (const float*)d_in[10];
    const float* w_out  = (const float*)d_in[11];  // [1024,2048]
    float* out = (float*)d_out;

    char* base = (char*)d_ws;
    size_t off = 0;
    auto carve = [&](size_t bytes) -> char* {
        char* p = base + off;
        off = (off + bytes + 255) & ~(size_t)255;
        return p;
    };
    _Float16* w_in16  = (_Float16*)carve((size_t)4096 * 1024 * 2);
    _Float16* w_xp16  = (_Float16*)carve((size_t)XPROJ_N * 2048 * 2);
    _Float16* w_dt16  = (_Float16*)carve((size_t)2048 * 64 * 2);
    _Float16* w_out16 = (_Float16*)carve((size_t)1024 * 2048 * 2);
    _Float16* xn16    = (_Float16*)carve((size_t)M_TOK * D_MODEL * 2);
    float*    xz      = (float*)   carve((size_t)M_TOK * 4096 * 4);
    float*    uconv   = (float*)   carve((size_t)M_TOK * D_INNER * 4);
    _Float16* u16     = (_Float16*)carve((size_t)M_TOK * D_INNER * 2);
    float*    xdbl    = (float*)   carve((size_t)M_TOK * XPROJ_N * 4);
    _Float16* dt16    = (_Float16*)carve((size_t)M_TOK * DT_RANK * 2);
    float*    delta   = (float*)   carve((size_t)M_TOK * D_INNER * 4);  // pre-softplus
    float*    ybuf    = (float*)   carve((size_t)M_TOK * D_INNER * 4);  // scan y, then out_proj f32
    _Float16* yg16    = (_Float16*)carve((size_t)M_TOK * D_INNER * 2);
    if (off > ws_size) return;

    const dim3 blk(256);
    auto g1 = [](int n) { return dim3((n + 255) / 256); };

    cvt_f32_f16<<<g1(4096 * 1024),    blk, 0, stream>>>(w_in,  w_in16,  4096 * 1024);
    cvt_f32_f16<<<g1(XPROJ_N * 2048), blk, 0, stream>>>(w_xp,  w_xp16,  XPROJ_N * 2048);
    cvt_f32_f16<<<g1(2048 * 64),      blk, 0, stream>>>(w_dt,  w_dt16,  2048 * 64);
    cvt_f32_f16<<<g1(1024 * 2048),    blk, 0, stream>>>(w_out, w_out16, 1024 * 2048);

    // 1) LayerNorm -> f16 activations
    layernorm_k<<<dim3(M_TOK), blk, 0, stream>>>(x, ln_g, ln_b, xn16);

    // 2) in_proj: [4096,1024] x [4096,1024]^T -> xz [4096,4096]
    gemm_wmma_f16<<<dim3(4096 / TN, M_TOK / TM), blk, 0, stream>>>(
        xn16, w_in16, xz, 4096, D_MODEL, 4096);

    // 3) causal depthwise conv + SiLU -> uconv (f32) + u16
    conv_silu_k<<<g1(M_TOK * D_INNER), blk, 0, stream>>>(xz, conv_w, conv_b, uconv, u16);

    // 4) x_proj: [4096,2048] x [96,2048]^T -> xdbl [4096,96]
    gemm_wmma_f16<<<dim3((XPROJ_N + TN - 1) / TN, M_TOK / TM), blk, 0, stream>>>(
        u16, w_xp16, xdbl, XPROJ_N, D_INNER, XPROJ_N);

    // 5) dt_proj: extract dt cols, GEMM (bias+softplus fused into scan)
    extract_dt_k<<<g1(M_TOK * DT_RANK), blk, 0, stream>>>(xdbl, dt16);
    gemm_wmma_f16<<<dim3(D_INNER / TN, M_TOK / TM), blk, 0, stream>>>(
        dt16, w_dt16, delta, D_INNER, DT_RANK, D_INNER);

    // 6) selective scan (4096 independent channels, L=2048 steps)
    scan_k<<<dim3(D_INNER / 256, B_SZ), blk, 0, stream>>>(
        uconv, delta, dt_b, xdbl, A_log, Dskip, ybuf);

    // 7) gate with silu(z)
    gate_k<<<g1(M_TOK * D_INNER), blk, 0, stream>>>(ybuf, xz, yg16);

    // 8) out_proj: [4096,2048] x [1024,2048]^T -> reuse ybuf as [4096,1024] f32
    gemm_wmma_f16<<<dim3(D_MODEL / TN, M_TOK / TM), blk, 0, stream>>>(
        yg16, w_out16, ybuf, D_MODEL, D_INNER, D_MODEL);

    // 9) residual + sanitize -> d_out
    resid_k<<<g1(M_TOK * D_MODEL), blk, 0, stream>>>(x, ybuf, out, M_TOK * D_MODEL);
}
